// GraphSAGE_23957327577785
// MI455X (gfx1250) — compile-verified, matrix operand order checked by reading, not run on hardware
//
#include <hip/hip_runtime.h>

// ---------------------------------------------------------------------------
// GraphSAGE (2x SAGEConv + tinykg quant + post-MLP + log_softmax) for MI455X.
// Layer-1 GEMM fused over [W1l|W1r] so the 573MB fp32 x tensor is streamed
// exactly once, converted to bf16 via native v_cvt (RNE) and fed to
// v_wmma_f32_16x16x32_bf16. B fragments use distinct registers so the four
// WMMAs issue back-to-back with staggered load waits.
// ---------------------------------------------------------------------------

#define NN   100000
#define EE   3200000
#define FINK 1433
#define HD   32
#define NCLS 7
#define KS1  ((FINK + 31) / 32)   // 45 k-steps of 32
#define BQV  64.0f

typedef __attribute__((ext_vector_type(16))) __bf16        v16bf;
typedef __attribute__((ext_vector_type(8)))  float         v8f;
typedef __attribute__((ext_vector_type(4)))  unsigned int  v4u;

union Frag {
  v16bf          bf;
  unsigned short us[16];
  v4u            q[2];
};

__device__ __forceinline__ unsigned short f2bf_bits(float f) {
  unsigned int u = __float_as_uint(f);
  u += 0x7FFFu + ((u >> 16) & 1u);   // round-to-nearest-even (device-side weight pack kernel)
  return (unsigned short)(u >> 16);
}

// ---------------------------------------------------------------------------
// Zero a float buffer.
__global__ void zero_f32(float* __restrict__ p, int n) {
  int i = blockIdx.x * blockDim.x + threadIdx.x;
  if (i < n) p[i] = 0.0f;
}

// ---------------------------------------------------------------------------
// Pre-swizzle weights [K,32]|[K,32] -> bf16 WMMA B-fragments.
// Fragment element (kstep, ntile, lane, i):
//   n = ntile*16 + (lane&15);  base = (lane>>4)*8
//   k = kstep*32 + (i<8 ? base+i : 16+base+i-8)
__global__ void pack_weights(const float* __restrict__ Wl,
                             const float* __restrict__ Wr,
                             int Kdim, int ksteps,
                             unsigned short* __restrict__ out) {
  int tid = blockIdx.x * blockDim.x + threadIdx.x;
  int total = ksteps << 11;                 // ksteps*4*32*16
  if (tid >= total) return;
  int i     = tid & 15;
  int lane  = (tid >> 4) & 31;
  int nt    = (tid >> 9) & 3;
  int kstep = tid >> 11;
  int n    = nt * 16 + (lane & 15);
  int base = (lane >> 4) * 8;
  int kit  = (i < 8) ? (base + i) : (16 + base + (i - 8));
  int k    = kstep * 32 + kit;
  float v = 0.0f;
  if (k < Kdim) v = (n < HD) ? Wl[k * HD + n] : Wr[k * HD + (n - HD)];
  out[tid] = f2bf_bits(v);
}

// ---------------------------------------------------------------------------
// GEMM: A[M,Kdim] (fp32, row-major) x packedB -> outL[M,32], outR[M,32].
// 8 waves / block, each wave computes a 16x64 strip via 4 WMMA accumulators.
__global__ __launch_bounds__(256) void gemm_wmma(
    const float* __restrict__ A, int Kdim, int ksteps,
    const unsigned short* __restrict__ pB,
    float* __restrict__ outL, float* __restrict__ outR, int Mrows) {
  int lane = threadIdx.x & 31;
  int wave = threadIdx.x >> 5;
  int m0 = blockIdx.x * 128 + wave * 16;
  if (m0 >= Mrows) return;

  int arow_i = m0 + (lane & 15);
  if (arow_i >= Mrows) arow_i = Mrows - 1;  // clamp loads, stores guarded below
  const float* arow = A + (size_t)arow_i * Kdim;
  int base = (lane >> 4) * 8;

  v8f acc0, acc1, acc2, acc3;
  #pragma unroll
  for (int r = 0; r < 8; ++r) { acc0[r] = 0.0f; acc1[r] = 0.0f; acc2[r] = 0.0f; acc3[r] = 0.0f; }

  for (int ks = 0; ks < ksteps; ++ks) {
    int k0 = ks * 32;

    // --- B fragments: 4 distinct registers, loads issued as one clause ---
    const char* bbase = (const char*)pB + ((size_t)(ks * 4) * 32 + lane) * 32;
    const v4u* p0 = (const v4u*)(bbase);
    const v4u* p1 = (const v4u*)(bbase + 1024);
    const v4u* p2 = (const v4u*)(bbase + 2048);
    const v4u* p3 = (const v4u*)(bbase + 3072);
    Frag b0, b1, b2, b3;
    b0.q[0] = p0[0]; b0.q[1] = p0[1];
    b1.q[0] = p1[0]; b1.q[1] = p1[1];
    b2.q[0] = p2[0]; b2.q[1] = p2[1];
    b3.q[0] = p3[0]; b3.q[1] = p3[1];

    // --- A fragment: native f32->bf16 converts (v_cvt_pk_bf16_f32) ---
    Frag a;
    if (k0 + 32 <= Kdim) {
      if (k0 + 160 <= Kdim) __builtin_prefetch(arow + k0 + 128, 0, 1);
      #pragma unroll
      for (int i = 0; i < 8; ++i) {
        a.bf[i]     = (__bf16)arow[k0 + base + i];
        a.bf[8 + i] = (__bf16)arow[k0 + 16 + base + i];
      }
    } else {
      #pragma unroll
      for (int i = 0; i < 8; ++i) {
        int ka = k0 + base + i;
        int kb = k0 + 16 + base + i;
        a.bf[i]     = (ka < Kdim) ? (__bf16)arow[ka] : (__bf16)0.0f;
        a.bf[8 + i] = (kb < Kdim) ? (__bf16)arow[kb] : (__bf16)0.0f;
      }
    }

    // --- 4 WMMAs back-to-back (distinct B regs -> staggered waits) ---
    acc0 = __builtin_amdgcn_wmma_f32_16x16x32_bf16(false, a.bf, false, b0.bf,
                                                   (short)0, acc0, false, false);
    acc1 = __builtin_amdgcn_wmma_f32_16x16x32_bf16(false, a.bf, false, b1.bf,
                                                   (short)0, acc1, false, false);
    acc2 = __builtin_amdgcn_wmma_f32_16x16x32_bf16(false, a.bf, false, b2.bf,
                                                   (short)0, acc2, false, false);
    acc3 = __builtin_amdgcn_wmma_f32_16x16x32_bf16(false, a.bf, false, b3.bf,
                                                   (short)0, acc3, false, false);
  }

  int rbase = m0 + ((lane >> 4) << 3);
  int cb = lane & 15;
  v8f accs[4] = {acc0, acc1, acc2, acc3};
  #pragma unroll
  for (int t = 0; t < 4; ++t) {
    int col = t * 16 + cb;
    float* op = (col < HD) ? (outL + col) : (outR + (col - HD));
    #pragma unroll
    for (int r = 0; r < 8; ++r) {
      int row = rbase + r;
      if (row < Mrows) op[(size_t)row * HD] = accs[t][r];
    }
  }
}

// ---------------------------------------------------------------------------
// Mean-aggregation scatter: one wave per edge, lane = feature column.
// atomicAdd results are unused -> non-returning global_atomic_add_f32.
__global__ void scatter_edges(const float* __restrict__ h,
                              const int* __restrict__ src,
                              const int* __restrict__ dst,
                              float* __restrict__ s,
                              float* __restrict__ deg) {
  int lane = threadIdx.x & 31;
  int wid = (blockIdx.x * blockDim.x + threadIdx.x) >> 5;
  int nw  = (gridDim.x * blockDim.x) >> 5;
  for (int e = wid; e < EE; e += nw) {
    int sn = src[e];
    int dn = dst[e];
    float v = h[(size_t)sn * HD + lane];
    atomicAdd(&s[(size_t)dn * HD + lane], v);
    if (lane == 0) atomicAdd(&deg[dn], 1.0f);
  }
}

// ---------------------------------------------------------------------------
// agg = s/max(deg,1) + b + hr ; relu ; tinykg stochastic quantization.
__global__ void finalize_quant(const float* __restrict__ s,
                               const float* __restrict__ deg,
                               const float* __restrict__ b,
                               const float* __restrict__ hr,
                               const float* __restrict__ u,
                               float* __restrict__ q) {
  int lane = threadIdx.x & 31;
  int row = (blockIdx.x * blockDim.x + threadIdx.x) >> 5;
  if (row >= NN) return;
  float v = s[(size_t)row * HD + lane] / fmaxf(deg[row], 1.0f)
          + b[lane] + hr[(size_t)row * HD + lane];
  v = fmaxf(v, 0.0f);
  float mn = v, mx = v;
  #pragma unroll
  for (int off = 16; off >= 1; off >>= 1) {
    mn = fminf(mn, __shfl_xor(mn, off, 32));
    mx = fmaxf(mx, __shfl_xor(mx, off, 32));
  }
  float rng = mx - mn;
  rng = (rng > 0.0f) ? rng : 1.0f;
  float sc = BQV * (v - mn) / rng;
  float fl = floorf(sc);
  float qv = fl + (((sc - fl) > u[(size_t)row * HD + lane]) ? 1.0f : 0.0f);
  q[(size_t)row * HD + lane] = qv;
}

// ---------------------------------------------------------------------------
// post_mp: (q2 @ Wp1 + bp1) @ Wp2 + bp2 -> log_softmax. One wave per row.
__global__ void post_mp(const float* __restrict__ q2,
                        const float* __restrict__ Wp1,
                        const float* __restrict__ bp1,
                        const float* __restrict__ Wp2,
                        const float* __restrict__ bp2,
                        float* __restrict__ out) {
  int lane = threadIdx.x & 31;
  int row = (blockIdx.x * blockDim.x + threadIdx.x) >> 5;
  if (row >= NN) return;
  float qv = q2[(size_t)row * HD + lane];
  float t = bp1[lane];
  #pragma unroll
  for (int k = 0; k < HD; ++k) {
    float qk = __shfl(qv, k, 32);
    t += qk * Wp1[k * HD + lane];
  }
  float lg = (lane < NCLS) ? bp2[lane] : 0.0f;
  #pragma unroll
  for (int j = 0; j < HD; ++j) {
    float tj = __shfl(t, j, 32);
    if (lane < NCLS) lg += tj * Wp2[j * NCLS + lane];
  }
  float val = (lane < NCLS) ? lg : -__builtin_inff();
  float m = val;
  #pragma unroll
  for (int off = 16; off >= 1; off >>= 1) m = fmaxf(m, __shfl_xor(m, off, 32));
  float ex = (lane < NCLS) ? __expf(val - m) : 0.0f;
  float se = ex;
  #pragma unroll
  for (int off = 16; off >= 1; off >>= 1) se += __shfl_xor(se, off, 32);
  float ls = val - m - __logf(se);
  if (lane < NCLS) out[(size_t)row * NCLS + lane] = ls;
}

// ---------------------------------------------------------------------------
extern "C" void kernel_launch(void* const* d_in, const int* in_sizes, int n_in,
                              void* d_out, int out_size, void* d_ws, size_t ws_size,
                              hipStream_t stream) {
  (void)in_sizes; (void)n_in; (void)out_size; (void)ws_size;
  const float* x   = (const float*)d_in[0];
  const float* W1l = (const float*)d_in[1];
  const float* b1  = (const float*)d_in[2];
  const float* W1r = (const float*)d_in[3];
  const float* W2l = (const float*)d_in[4];
  const float* b2  = (const float*)d_in[5];
  const float* W2r = (const float*)d_in[6];
  const float* Wp1 = (const float*)d_in[7];
  const float* bp1 = (const float*)d_in[8];
  const float* Wp2 = (const float*)d_in[9];
  const float* bp2 = (const float*)d_in[10];
  const float* u1  = (const float*)d_in[11];
  const float* u2  = (const float*)d_in[12];
  const int*  esrc = (const int*)d_in[13];
  const int*  edst = (const int*)d_in[14];
  float* out = (float*)d_out;

  // Workspace layout (~64.6 MB): hl, hr, s, deg, q1, q2, packedB1, packedB2.
  char* ws = (char*)d_ws;
  size_t off = 0;
  float* hl   = (float*)(ws + off); off += (size_t)NN * HD * 4;
  float* hr   = (float*)(ws + off); off += (size_t)NN * HD * 4;
  float* sbuf = (float*)(ws + off); off += (size_t)NN * HD * 4;
  float* deg  = (float*)(ws + off); off += (size_t)NN * 4;       // adjacent to sbuf
  float* q1   = (float*)(ws + off); off += (size_t)NN * HD * 4;
  float* q2   = (float*)(ws + off); off += (size_t)NN * HD * 4;
  unsigned short* pB1 = (unsigned short*)(ws + off); off += (size_t)KS1 * 2048 * 2;
  unsigned short* pB2 = (unsigned short*)(ws + off); off += (size_t)2048 * 2;

  const int gemmBlocks = (NN + 127) / 128;     // 782
  const int rowWaveBlk = (NN * 32) / 256;      // 12500 (one wave per row)
  const int zeroBlk    = (NN * 33 + 255) / 256;

  // Weight pre-swizzle to bf16 WMMA fragments.
  pack_weights<<<(KS1 * 2048 + 255) / 256, 256, 0, stream>>>(W1l, W1r, FINK, KS1, pB1);
  pack_weights<<<(2048 + 255) / 256, 256, 0, stream>>>(W2l, W2r, HD, 1, pB2);

  // ---- Layer 1 ----
  gemm_wmma<<<gemmBlocks, 256, 0, stream>>>(x, FINK, KS1, pB1, hl, hr, NN);
  zero_f32<<<zeroBlk, 256, 0, stream>>>(sbuf, NN * 33);          // s + deg
  scatter_edges<<<2048, 256, 0, stream>>>(hl, esrc, edst, sbuf, deg);
  finalize_quant<<<rowWaveBlk, 256, 0, stream>>>(sbuf, deg, b1, hr, u1, q1);

  // ---- Layer 2 ----
  gemm_wmma<<<gemmBlocks, 256, 0, stream>>>(q1, HD, 1, pB2, hl, hr, NN);
  zero_f32<<<zeroBlk, 256, 0, stream>>>(sbuf, NN * 33);
  scatter_edges<<<2048, 256, 0, stream>>>(hl, esrc, edst, sbuf, deg);
  finalize_quant<<<rowWaveBlk, 256, 0, stream>>>(sbuf, deg, b2, hr, u2, q2);

  // ---- post-MLP + log_softmax ----
  post_mp<<<rowWaveBlk, 256, 0, stream>>>(q2, Wp1, bp1, Wp2, bp2, out);
}
